// Tno_25391846654067
// MI455X (gfx1250) — matrix operand order repaired and checked by reading, not compiled, not run on hardware
//
#include <hip/hip_runtime.h>
#include <hip/hip_bf16.h>
#include <cmath>

// ---------------------------------------------------------------------------
// TNN Toeplitz operator for MI455X (gfx1250), wave32 + WMMA bf16.
//   H=8, D=64, RPE_DIM=64, LAYERS=3, b=8, n=4096, gamma=0.999
// out[b,h,i,d] = sum_{j<n} a[h,(i-j) mod 2n, d] * x[b,h,j,d]
// ---------------------------------------------------------------------------

typedef __bf16 bf16;
typedef __attribute__((ext_vector_type(16))) __bf16 v16bf;
typedef __attribute__((ext_vector_type(8)))  float  v8f;

#define NSEQ   4096
#define TWO_N  8192
#define HH     8
#define DD     64
#define BB     8
#define RPE    64

// ---------------------------------------------------------------------------
// Kernel 1: RPE MLP -> a_bf[h][d][t] (bf16), t in [0, 2n), decay applied.
// ---------------------------------------------------------------------------
__global__ __launch_bounds__(64) void rpe_mlp_kernel(
    const float* __restrict__ pos_w, const float* __restrict__ pos_b,
    const float* __restrict__ lw,    const float* __restrict__ lb,
    const float* __restrict__ ow,    const float* __restrict__ ob,
    bf16* __restrict__ a_bf)
{
    __shared__ float sh_y[RPE];
    __shared__ float red[RPE];
    const int r = blockIdx.x;          // table row 0..8191
    const int t = threadIdx.x;         // 0..63

    float off;
    if (r == 0 || r == NSEQ)      off = 0.0f;
    else if (r < NSEQ)            off = (float)r;
    else                          off = (float)(r - TWO_N);

    float xv = off * pos_w[t] + pos_b[t];

    #pragma unroll
    for (int L = 0; L < 3; ++L) {
        red[t] = xv * xv;
        __syncthreads();
        for (int s = 32; s > 0; s >>= 1) {
            if (t < s) red[t] += red[t + s];
            __syncthreads();
        }
        float rms = sqrtf(red[0]) * 0.125f;
        __syncthreads();
        float y = fmaxf(xv / (rms + 1e-8f), 0.0f);
        sh_y[t] = y;
        __syncthreads();
        float acc = lb[L * RPE + t];
        const float* W = lw + L * RPE * RPE;
        #pragma unroll 8
        for (int s = 0; s < RPE; ++s) acc += sh_y[s] * W[s * RPE + t];
        xv = acc;
        __syncthreads();
    }

    red[t] = xv * xv;
    __syncthreads();
    for (int s = 32; s > 0; s >>= 1) {
        if (t < s) red[t] += red[t + s];
        __syncthreads();
    }
    float rms = sqrtf(red[0]) * 0.125f;
    __syncthreads();
    float y = fmaxf(xv / (rms + 1e-8f), 0.0f);
    sh_y[t] = y;
    __syncthreads();

    const float decay = powf(0.999f, fabsf(off));
    #pragma unroll
    for (int u = 0; u < HH; ++u) {
        const int c = u * DD + t;
        float acc = ob[c];
        #pragma unroll 8
        for (int s = 0; s < RPE; ++s) acc += sh_y[s] * ow[s * (HH * DD) + c];
        a_bf[((size_t)(u * DD + t)) * TWO_N + r] = (bf16)(acc * decay);
    }
}

// ---------------------------------------------------------------------------
// Kernel 2: x[b][h][j][d] f32 -> xT[h][d][j][b] bf16 (16B row per (h,d,j)).
// ---------------------------------------------------------------------------
struct alignas(16) Bf8 { bf16 v[8]; };

__global__ __launch_bounds__(256) void transpose_x_kernel(
    const float* __restrict__ x, bf16* __restrict__ xT)
{
    const int g = blockIdx.x * 256 + threadIdx.x;
    const int j = g & (NSEQ - 1);
    const int d = (g >> 12) & (DD - 1);
    const int h = g >> 18;
    Bf8 p;
    #pragma unroll
    for (int b = 0; b < BB; ++b)
        p.v[b] = (bf16)x[(((size_t)(b * HH + h)) * NSEQ + j) * DD + d];
    *(Bf8*)(xT + (((size_t)(h * DD + d)) * NSEQ + j) * BB) = p;
}

// ---------------------------------------------------------------------------
// Kernel 3: Toeplitz matvec as WMMA bf16 GEMM.
// Block = 256 threads = 8 waves, one (h,d) per block; wave w owns 128 rows of
// i (8 accumulator tiles). K chunked by 128. Toeplitz strip of `a` is staged
// REVERSED in LDS so fragments are ascending-contiguous (pure ds_load_b128,
// no half-swaps). Fragment reuse: frag(it,ks)==frag(it-2,ks-1), so only tiles
// 0,1 load fresh A per 32-K step; others rotate in registers (period-4 role
// rotation over the fully unrolled 4-step chunk => zero register copies).
// ---------------------------------------------------------------------------
__global__ __launch_bounds__(256) void toeplitz_wmma_kernel(
    const bf16* __restrict__ a_bf,   // [H*D][2n]
    const bf16* __restrict__ xT,     // [H*D][n][8]
    float* __restrict__ out)         // [b][H][n][D]
{
    __shared__ bf16 b_sh[128 * 8];       // K-chunk of x (128 rows x 8 batch)
    __shared__ bf16 a_sh[8 * 256];       // per-wave reversed Toeplitz strips

    const int tid  = threadIdx.x;
    const int wave = tid >> 5;
    const int lane = tid & 31;
    const int hd   = blockIdx.x >> 2;    // 0..511
    const int iblk = blockIdx.x & 3;
    const int h    = hd >> 6;
    const int d    = hd & 63;

    const bf16* a_hd = a_bf + (size_t)hd * TWO_N;
    const bf16* x_hd = xT   + (size_t)hd * (NSEQ * BB);
    const int Ibase  = iblk * 1024 + wave * 128;

    const int m   = lane & 15;           // A row within tile
    const int khi = (lane >> 4) << 3;    // 0 or 8 (A K-group per lane half)
    const bf16* rev = a_sh + wave * 256;

    v8f acc[8];
    #pragma unroll
    for (int it = 0; it < 8; ++it)
        #pragma unroll
        for (int e = 0; e < 8; ++e) acc[it][e] = 0.0f;

    // A fragment from reversed strip: two ascending contiguous 8-half runs.
    auto FRAG = [&](int it, int ks) -> v16bf {
        const bf16* p = rev + (128 + ks * 32 - it * 16 - m + khi);
        v16bf f;
        #pragma unroll
        for (int e = 0; e < 8; ++e) f[e]     = p[e];
        #pragma unroll
        for (int e = 0; e < 8; ++e) f[8 + e] = p[16 + e];
        return f;
    };
    // B fragment: lane&15 = N col (b<8 live, rest 0); elements = K rows (table order).
    auto BFRAG = [&](int ks) -> v16bf {
        const int n = lane & 15;
        const bf16* p = b_sh + (ks * 32 + khi) * 8 + (n & 7);
        v16bf f;
        #pragma unroll
        for (int e = 0; e < 8; ++e) f[e]     = (n < 8) ? p[e * 8] : (bf16)0.0f;
        #pragma unroll
        for (int e = 0; e < 8; ++e) f[8 + e] = (n < 8) ? p[(16 + e) * 8] : (bf16)0.0f;
        return f;
    };

#define WMMA_ACC(I, A, BF) \
    acc[I] = __builtin_amdgcn_wmma_f32_16x16x32_bf16(false, A, false, BF, (short)0, acc[I], false, false)

#define STEP(KS, T0, T1, T2, T3, T4, T5, T6, T7)   \
    {                                              \
        v16bf bfr = BFRAG(KS);                     \
        T0 = FRAG(0, KS);                          \
        T1 = FRAG(1, KS);                          \
        WMMA_ACC(0, T0, bfr);                      \
        WMMA_ACC(1, T1, bfr);                      \
        WMMA_ACC(2, T2, bfr);                      \
        WMMA_ACC(3, T3, bfr);                      \
        WMMA_ACC(4, T4, bfr);                      \
        WMMA_ACC(5, T5, bfr);                      \
        WMMA_ACC(6, T6, bfr);                      \
        WMMA_ACC(7, T7, bfr);                      \
    }

    v16bf F0, F1, F2, F3, F4, F5, F6, F7;
    #pragma unroll
    for (int e = 0; e < 16; ++e) {
        F0[e] = (bf16)0.0f; F1[e] = (bf16)0.0f; F2[e] = (bf16)0.0f;
        F3[e] = (bf16)0.0f; F4[e] = (bf16)0.0f; F5[e] = (bf16)0.0f;
        F6[e] = (bf16)0.0f; F7[e] = (bf16)0.0f;
    }

    for (int Kb = 0; Kb < NSEQ; Kb += 128) {
        __syncthreads();
        // stage B chunk: 128 rows x 16 bytes
        if (tid < 128)
            *(uint4*)(b_sh + tid * 8) =
                *(const uint4*)(x_hd + (size_t)(Kb + tid) * 8);
        // stage reversed per-wave Toeplitz strip:
        //   rev[q] = a[(tmax - q) mod 2n], tmax = Ibase - Kb + 128
        const int tmax = Ibase - Kb + 128;
        #pragma unroll
        for (int q = lane; q < 256; q += 32)
            a_sh[wave * 256 + q] = a_hd[(tmax - q) & (TWO_N - 1)];
        __syncthreads();

        if (Kb == 0) {   // prologue: seed carried fragments (tiles 2..7, ks=0)
            F2 = FRAG(2, 0); F3 = FRAG(3, 0); F4 = FRAG(4, 0);
            F5 = FRAG(5, 0); F6 = FRAG(6, 0); F7 = FRAG(7, 0);
        }
        STEP(0, F0, F1, F2, F3, F4, F5, F6, F7)
        STEP(1, F6, F7, F0, F1, F2, F3, F4, F5)
        STEP(2, F4, F5, F6, F7, F0, F1, F2, F3)
        STEP(3, F2, F3, F4, F5, F6, F7, F0, F1)
    }

    // store: VGPR r holds M = r (lanes 0-15) / 8+r (lanes 16-31), lane&15 = N
    const int ncol = lane & 15;
    if (ncol < 8) {
        const int b   = ncol;
        const int mhi = (lane >> 4) * 8;
        #pragma unroll
        for (int it = 0; it < 8; ++it)
            #pragma unroll
            for (int r = 0; r < 8; ++r) {
                const int i = Ibase + it * 16 + mhi + r;
                __builtin_nontemporal_store(
                    acc[it][r],
                    &out[(((size_t)(b * HH + h)) * NSEQ + i) * DD + d]);
            }
    }
#undef STEP
#undef WMMA_ACC
}

// ---------------------------------------------------------------------------
extern "C" void kernel_launch(void* const* d_in, const int* in_sizes, int n_in,
                              void* d_out, int out_size, void* d_ws, size_t ws_size,
                              hipStream_t stream) {
    const float* x     = (const float*)d_in[0];
    const float* pos_w = (const float*)d_in[1];
    const float* pos_b = (const float*)d_in[2];
    const float* lw    = (const float*)d_in[3];
    const float* lb    = (const float*)d_in[4];
    const float* ow    = (const float*)d_in[5];
    const float* ob    = (const float*)d_in[6];
    float* out = (float*)d_out;

    bf16* a_bf = (bf16*)d_ws;                                    // 8 MB
    bf16* xT   = (bf16*)((char*)d_ws + (size_t)8 * 1024 * 1024); // 32 MB

    rpe_mlp_kernel<<<TWO_N, 64, 0, stream>>>(pos_w, pos_b, lw, lb, ow, ob, a_bf);
    transpose_x_kernel<<<(HH * DD * NSEQ) / 256, 256, 0, stream>>>(x, xT);
    toeplitz_wmma_kernel<<<HH * DD * 4, 256, 0, stream>>>(a_bf, xT, out);

    (void)in_sizes; (void)n_in; (void)out_size; (void)ws_size;
}
